// NeedlemanWunschAligner_15487652069568
// MI455X (gfx1250) — matrix operand order repaired and checked by reading, not compiled, not run on hardware
//
#include <hip/hip_runtime.h>
#include <hip/hip_bf16.h>
#include <math.h>

// ---------------- WMMA fragment types (gfx1250, wave32) ----------------
typedef __bf16 bf16_t;
typedef __attribute__((ext_vector_type(16))) __bf16 bf16x16;
typedef __attribute__((ext_vector_type(8)))  float  f32x8;

union FragAB { bf16x16 v; bf16_t h[16]; uint4 q[2]; };
union FragC  { f32x8  v;  float  f[8]; };

// A-matrix 16x32 bf16 layout (per ISA): lane m=lane&15 holds row m;
// lanes<16: K = {kb..kb+7, kb+16..kb+23}; lanes>=16: +8.
__device__ __forceinline__ bf16x16 load_frag_A(const bf16_t* A, int lda,
                                               int mBase, int kb, int lane) {
  const bf16_t* p = A + (size_t)(mBase + (lane & 15)) * lda + kb + ((lane >> 4) << 3);
  FragAB f;
  f.q[0] = *reinterpret_cast<const uint4*>(p);       // K = +0..7
  f.q[1] = *reinterpret_cast<const uint4*>(p + 16);  // K = +16..23
  return f.v;
}

// Same fragment shape but sourced from an LDS-staged tile with leading dim 128.
__device__ __forceinline__ bf16x16 load_frag_A_lds(const bf16_t* base,
                                                   int mBase, int kb2, int lane) {
  const bf16_t* p = base + (size_t)(mBase + (lane & 15)) * 128 + kb2 + ((lane >> 4) << 3);
  FragAB f;
  f.q[0] = *reinterpret_cast<const uint4*>(p);
  f.q[1] = *reinterpret_cast<const uint4*>(p + 16);
  return f.v;
}

// B-matrix 32x16 bf16, stored N-major (Bt is N x K row-major):
// lane n=lane&15 holds column n; lanes<16: K = {kb..kb+7, kb+16..kb+23}; lanes>=16: +8.
__device__ __forceinline__ bf16x16 load_frag_B(const bf16_t* Bt, int ldb,
                                               int nBase, int kb, int lane) {
  const bf16_t* p = Bt + (size_t)(nBase + (lane & 15)) * ldb + kb + ((lane >> 4) << 3);
  FragAB f;
  f.q[0] = *reinterpret_cast<const uint4*>(p);
  f.q[1] = *reinterpret_cast<const uint4*>(p + 16);
  return f.v;
}

__device__ __forceinline__ f32x8 wmma_bf16(bf16x16 a, bf16x16 b, f32x8 c) {
  return __builtin_amdgcn_wmma_f32_16x16x32_bf16(false, a, false, b, (short)0, c,
                                                 false, false);
}

__device__ __forceinline__ float sigmoidf(float x) {
  return 1.0f / (1.0f + expf(-x));
}

// CDNA5 async memory->LDS copy (16B per lane), tracked by ASYNCcnt.
__device__ __forceinline__ void async_copy_b128(unsigned int ldsAddr, const void* gaddr) {
  asm volatile("global_load_async_to_lds_b128 %0, %1, off"
               :: "v"(ldsAddr), "v"(gaddr) : "memory");
}
__device__ __forceinline__ void wait_asynccnt0() {
  asm volatile("s_wait_asynccnt 0x0" ::: "memory");
}

// ---------------- helper conversion kernels ----------------
// W: K x N f32  ->  Wt: N x K bf16 (N-major for WMMA B operand)
__global__ void k_convert_transpose(const float* __restrict__ W, bf16_t* __restrict__ Wt,
                                    int K, int N) {
  int idx = blockIdx.x * blockDim.x + threadIdx.x;
  if (idx >= N * K) return;
  int n = idx / K, k = idx % K;
  Wt[idx] = (bf16_t)W[(size_t)k * N + n];
}

__global__ void k_convert(const float* __restrict__ src, bf16_t* __restrict__ dst, int n) {
  int idx = blockIdx.x * blockDim.x + threadIdx.x;
  if (idx < n) dst[idx] = (bf16_t)src[idx];
}

__global__ void k_zero_state(bf16_t* __restrict__ hbuf, float* __restrict__ c) {
  int idx = blockIdx.x * blockDim.x + threadIdx.x;
  if (idx < 2 * 64 * 512) hbuf[idx] = (bf16_t)0.0f;
  if (idx < 64 * 512) c[idx] = 0.0f;
}

// ---------------- pre = gather(emb, tok) @ W_ih + b_lstm ----------------
// M = 32768 rows ordered (seq, batch, t); N = 2048; K = 512.
// grid (M/32, N/64), block 256 (8 waves, 2x4 tiles of 16x16).
__global__ void k_gemm_pre(const int* __restrict__ xTok, const int* __restrict__ yTok,
                           const bf16_t* __restrict__ emb, const bf16_t* __restrict__ WtIh,
                           const float* __restrict__ bLstm, float* __restrict__ pre) {
  const int lane = threadIdx.x & 31, wave = threadIdx.x >> 5;
  const int mBase = blockIdx.x * 32 + (wave >> 2) * 16;
  const int nBase = blockIdx.y * 64 + (wave & 3) * 16;

  // gather: A row = emb[token]
  int r = mBase + (lane & 15);
  int s = r >> 14, rem = r & 16383;
  int tok = s ? yTok[rem] : xTok[rem];
  const bf16_t* arow = emb + (size_t)tok * 512;

  f32x8 acc = {};
  for (int kb = 0; kb < 512; kb += 32) {
    FragAB fa;
    const bf16_t* p = arow + kb + ((lane >> 4) << 3);
    fa.q[0] = *reinterpret_cast<const uint4*>(p);
    fa.q[1] = *reinterpret_cast<const uint4*>(p + 16);
    bf16x16 b = load_frag_B(WtIh, 512, nBase, kb, lane);
    acc = wmma_bf16(fa.v, b, acc);
  }
  FragC fc; fc.v = acc;
  int col = nBase + (lane & 15);
  int rowBase = mBase + ((lane >> 4) << 3);
#pragma unroll
  for (int rr = 0; rr < 8; ++rr)
    pre[(size_t)(rowBase + rr) * 2048 + col] = fc.f[rr] + bLstm[col];
}

// ---------------- one LSTM timestep ----------------
// z = pre[:, t, :] + h @ W_hh ; gates; update c, h. Batch M = 64 (x rows 0..31, y 32..63).
// grid: 16 workgroups, each owns 32 hidden columns. Each wave owns one
// (16-row, 16-col) slice and computes all 4 gate tiles for it, so the
// nonlinearity + state update is pure per-lane work (no LDS staging of z).
// h is staged through LDS in double-buffered async K-chunks (ASYNCcnt pipeline).
__global__ void k_lstm_step(const float* __restrict__ pre, const bf16_t* __restrict__ WtHh,
                            const bf16_t* __restrict__ hIn, bf16_t* __restrict__ hOut,
                            float* __restrict__ c, bf16_t* __restrict__ hs, int t) {
  __shared__ bf16_t hstage[2][64][128];   // 2 x 16 KB double-buffered K-chunks of h
  const int wg = blockIdx.x;
  const int tid = threadIdx.x;
  const int lane = tid & 31, wave = tid >> 5;
  const int rt = wave >> 1;               // row-tile 0..3
  const int sub = wave & 1;               // col sub-tile 0..1
  const int colBase = wg * 32 + sub * 16;

  f32x8 accI = {}, accF = {}, accG = {}, accO = {};

  // stage one 64x128 bf16 chunk of h into LDS: 1024 x b128 over 256 threads
  auto stage = [&](int ck, int bufSel) {
#pragma unroll
    for (int j = 0; j < 4; ++j) {
      int idx = tid + j * 256;
      int row = idx >> 4, seg = idx & 15;
      unsigned int lds = (unsigned int)(uintptr_t)&hstage[bufSel][row][seg * 8];
      const bf16_t* g = hIn + (size_t)row * 512 + ck * 128 + seg * 8;
      async_copy_b128(lds, (const void*)g);
    }
  };

  stage(0, 0);
  for (int ck = 0; ck < 4; ++ck) {
    wait_asynccnt0();
    __syncthreads();                       // chunk ck landed for all waves
    if (ck + 1 < 4) stage(ck + 1, (ck + 1) & 1);
    const bf16_t* abase = &hstage[ck & 1][0][0];
    for (int kb2 = 0; kb2 < 128; kb2 += 32) {
      int kb = ck * 128 + kb2;
      bf16x16 a = wmma_bf16 == nullptr ? bf16x16{} : load_frag_A_lds(abase, rt * 16, kb2, lane);
      accI = wmma_bf16(a, load_frag_B(WtHh, 512,           colBase, kb, lane), accI);
      accF = wmma_bf16(a, load_frag_B(WtHh, 512,  512 + colBase, kb, lane), accF);
      accG = wmma_bf16(a, load_frag_B(WtHh, 512, 1024 + colBase, kb, lane), accG);
      accO = wmma_bf16(a, load_frag_B(WtHh, 512, 1536 + colBase, kb, lane), accO);
    }
  }

  // per-lane gate epilogue: i/f/g/o for (row, col) live on the same lane
  FragC fI, fF, fG, fO;
  fI.v = accI; fF.v = accF; fG.v = accG; fO.v = accO;
  int col = colBase + (lane & 15);
  int rowBase = rt * 16 + ((lane >> 4) << 3);
#pragma unroll
  for (int rr = 0; rr < 8; ++rr) {
    int row = rowBase + rr;
    int s = row >> 5, bb = row & 31;
    size_t preRow = (size_t)s * 16384 + (size_t)bb * 512 + t;
    const float* pr = pre + preRow * 2048;
    float zi = fI.f[rr] + pr[col];
    float zf = fF.f[rr] + pr[512 + col];
    float zg = fG.f[rr] + pr[1024 + col];
    float zo = fO.f[rr] + pr[1536 + col];
    float cOld = c[(size_t)row * 512 + col];
    float cNew = sigmoidf(zf) * cOld + sigmoidf(zi) * tanhf(zg);
    float hNew = sigmoidf(zo) * tanhf(cNew);
    c[(size_t)row * 512 + col] = cNew;
    hOut[(size_t)row * 512 + col] = (bf16_t)hNew;
    hs[preRow * 512 + col] = (bf16_t)hNew;   // row order matches pre: (seq, batch, t)
  }
}

// ---------------- z = hs @ W_out + b_out (output bf16) ----------------
// M = 32768, N = 512, K = 512. grid (1024, 8), block 256.
__global__ void k_gemm_out(const bf16_t* __restrict__ hs, const bf16_t* __restrict__ WtOut,
                           const float* __restrict__ bOut, bf16_t* __restrict__ z) {
  const int lane = threadIdx.x & 31, wave = threadIdx.x >> 5;
  const int mBase = blockIdx.x * 32 + (wave >> 2) * 16;
  const int nBase = blockIdx.y * 64 + (wave & 3) * 16;
  f32x8 acc = {};
  for (int kb = 0; kb < 512; kb += 32) {
    bf16x16 a = load_frag_A(hs, 512, mBase, kb, lane);
    bf16x16 b = load_frag_B(WtOut, 512, nBase, kb, lane);
    acc = wmma_bf16(a, b, acc);
  }
  FragC fc; fc.v = acc;
  int col = nBase + (lane & 15);
  int rowBase = mBase + ((lane >> 4) << 3);
#pragma unroll
  for (int rr = 0; rr < 8; ++rr)
    z[(size_t)(rowBase + rr) * 512 + col] = (bf16_t)(fc.f[rr] + bOut[col]);
}

// ---------------- gap score A[b] = concat(mean zx, mean zy) @ W_gap + b_gap ----------------
__global__ void k_gap(const bf16_t* __restrict__ z, const float* __restrict__ Wgap,
                      const float* __restrict__ bgap, float* __restrict__ Agap) {
  __shared__ float red[256];
  int b = blockIdx.x, tid = threadIdx.x;
  float part = 0.0f;
  for (int d = tid; d < 512; d += blockDim.x) {
    float sx = 0.0f, sy = 0.0f;
    for (int t = 0; t < 512; ++t) {
      sx += (float)z[((size_t)b * 512 + t) * 512 + d];
      sy += (float)z[((size_t)(16384 + b * 512) + t) * 512 + d];
    }
    part += (sx * (1.0f / 512.0f)) * Wgap[d] + (sy * (1.0f / 512.0f)) * Wgap[512 + d];
  }
  red[tid] = part;
  __syncthreads();
  for (int off = 128; off > 0; off >>= 1) {
    if (tid < off) red[tid] += red[tid + off];
    __syncthreads();
  }
  if (tid == 0) Agap[b] = red[0] + bgap[0];
}

// ---------------- theta[b] = zx_b @ zy_b^T ----------------
// per batch: M = N = K = 512. grid (16, 8, 32), block 256.
__global__ void k_gemm_theta(const bf16_t* __restrict__ z, float* __restrict__ theta) {
  const int b = blockIdx.z;
  const int lane = threadIdx.x & 31, wave = threadIdx.x >> 5;
  const int mBase = blockIdx.x * 32 + (wave >> 2) * 16;
  const int nBase = blockIdx.y * 64 + (wave & 3) * 16;
  const bf16_t* zx = z + (size_t)b * 512 * 512;
  const bf16_t* zy = z + (size_t)(16384 + b * 512) * 512;
  f32x8 acc = {};
  for (int kb = 0; kb < 512; kb += 32) {
    bf16x16 a  = load_frag_A(zx, 512, mBase, kb, lane);
    bf16x16 bm = load_frag_B(zy, 512, nBase, kb, lane);  // zy rows are N-major for A@B^T
    acc = wmma_bf16(a, bm, acc);
  }
  FragC fc; fc.v = acc;
  int col = nBase + (lane & 15);
  int rowBase = mBase + ((lane >> 4) << 3);
  float* th = theta + (size_t)b * 512 * 512;
#pragma unroll
  for (int rr = 0; rr < 8; ++rr)
    th[(size_t)(rowBase + rr) * 512 + col] = fc.f[rr];
}

// ---------------- Needleman-Wunsch soft-DP (anti-diagonal wavefront) ----------------
// one workgroup per batch; 3 rotating rows of 513 floats in LDS; 1023 barrier steps.
__global__ void k_nw(const float* __restrict__ theta, const float* __restrict__ Agap,
                     float* __restrict__ out) {
  const int N = 512, M = 512;
  __shared__ float buf[3][513];
  const int b = blockIdx.x, tid = threadIdx.x;
  const float A = Agap[b];
  const float NEGV = -1000000000.0f;
  for (int ii = tid; ii <= N; ii += blockDim.x) {
    buf[0][ii] = (ii == 0) ? 0.0f : NEGV;
    buf[1][ii] = NEGV;
  }
  __syncthreads();
  int p2 = 0, p1 = 1, pc = 2;
  const float* th = theta + (size_t)b * N * M;
  for (int k = 2; k <= N + M; ++k) {
    for (int ii = tid; ii <= N; ii += blockDim.x) {
      int jj = k - ii;
      float v;
      if (ii >= 1 && jj >= 1 && jj <= M) {
        float td = th[(size_t)(ii - 1) * M + (jj - 1)];
        float a1 = buf[p1][ii - 1] + A;   // shift(p1) + A
        float a2 = buf[p2][ii - 1];       // shift(p2)
        float a3 = buf[p1][ii] + A;       // p1 + A
        float m = fmaxf(a1, fmaxf(a2, a3));
        float lse = m + logf(expf(a1 - m) + expf(a2 - m) + expf(a3 - m));
        v = td + lse;
      } else {
        v = NEGV;
      }
      buf[pc][ii] = v;
    }
    __syncthreads();
    int tmp = p2; p2 = p1; p1 = pc; pc = tmp;
  }
  if (tid == 0) out[b] = buf[p1][N];
}

// ---------------- host orchestration ----------------
extern "C" void kernel_launch(void* const* d_in, const int* in_sizes, int n_in,
                              void* d_out, int out_size, void* d_ws, size_t ws_size,
                              hipStream_t stream) {
  (void)in_sizes; (void)n_in; (void)out_size; (void)ws_size;
  const int*   x      = (const int*)d_in[0];
  const int*   y      = (const int*)d_in[1];
  const float* emb    = (const float*)d_in[2];
  const float* W_ih   = (const float*)d_in[3];
  const float* W_hh   = (const float*)d_in[4];
  const float* b_lstm = (const float*)d_in[5];
  const float* W_out  = (const float*)d_in[6];
  const float* b_out  = (const float*)d_in[7];
  const float* W_gap  = (const float*)d_in[8];
  const float* b_gap  = (const float*)d_in[9];
  float* out = (float*)d_out;

  char* ws = (char*)d_ws;
  size_t off = 0;
  auto alloc = [&](size_t bytes) -> void* {
    void* p = ws + off;
    off = (off + bytes + 255) & ~(size_t)255;
    return p;
  };
  bf16_t* WtIh  = (bf16_t*)alloc((size_t)2048 * 512 * 2);
  bf16_t* WtHh  = (bf16_t*)alloc((size_t)2048 * 512 * 2);
  bf16_t* WtOut = (bf16_t*)alloc((size_t)512 * 512 * 2);
  bf16_t* embBf = (bf16_t*)alloc((size_t)22 * 512 * 2);
  float*  pre   = (float*)alloc((size_t)32768 * 2048 * 4);
  bf16_t* hbuf  = (bf16_t*)alloc((size_t)2 * 64 * 512 * 2);   // double-buffered h
  float*  cbuf  = (float*)alloc((size_t)64 * 512 * 4);
  bf16_t* hs    = (bf16_t*)alloc((size_t)32768 * 512 * 2);
  bf16_t* zbf   = (bf16_t*)alloc((size_t)32768 * 512 * 2);
  float*  theta = (float*)alloc((size_t)32 * 512 * 512 * 4);
  float*  Agap  = (float*)alloc((size_t)32 * 4);

  // 1. weight conversion / transposition to bf16 N-major
  k_convert_transpose<<<(2048 * 512 + 255) / 256, 256, 0, stream>>>(W_ih, WtIh, 512, 2048);
  k_convert_transpose<<<(2048 * 512 + 255) / 256, 256, 0, stream>>>(W_hh, WtHh, 512, 2048);
  k_convert_transpose<<<(512 * 512 + 255) / 256, 256, 0, stream>>>(W_out, WtOut, 512, 512);
  k_convert<<<(22 * 512 + 255) / 256, 256, 0, stream>>>(emb, embBf, 22 * 512);

  // 2. pre = emb[tok] @ W_ih + b (both sequences, M = 32768)
  k_gemm_pre<<<dim3(1024, 32), 256, 0, stream>>>(x, y, embBf, WtIh, b_lstm, pre);

  // 3. LSTM scan, 512 sequential steps, h double-buffered across launches
  k_zero_state<<<(2 * 64 * 512 + 255) / 256, 256, 0, stream>>>(hbuf, cbuf);
  for (int t = 0; t < 512; ++t) {
    bf16_t* hIn  = hbuf + (size_t)(t & 1) * 64 * 512;
    bf16_t* hOut = hbuf + (size_t)((t + 1) & 1) * 64 * 512;
    k_lstm_step<<<16, 256, 0, stream>>>(pre, WtHh, hIn, hOut, cbuf, hs, t);
  }

  // 4. z = hs @ W_out + b_out (bf16 output for theta + means)
  k_gemm_out<<<dim3(1024, 8), 256, 0, stream>>>(hs, WtOut, b_out, zbf);

  // 5. gap scores
  k_gap<<<32, 256, 0, stream>>>(zbf, W_gap, b_gap, Agap);

  // 6. theta = zx @ zy^T per batch
  k_gemm_theta<<<dim3(16, 8, 32), 256, 0, stream>>>(zbf, theta);

  // 7. soft Needleman-Wunsch DP
  k_nw<<<32, 512, 0, stream>>>(theta, Agap, out);
}
